// AttentionHead_41532333752721
// MI455X (gfx1250) — matrix operand order repaired and checked by reading, not compiled, not run on hardware
//
#include <hip/hip_runtime.h>

typedef __attribute__((ext_vector_type(16))) _Float16 v16h;
typedef __attribute__((ext_vector_type(8)))  _Float16 v8h;
typedef __attribute__((ext_vector_type(8)))  float    v8f;
typedef __attribute__((ext_vector_type(4)))  float    v4f;

#define SEQ   2048
#define NHEAD 8
#define EMB   512
#define HD    64
#define NB    4

#define CHUNK_HALVES (12 * 32 * 16)          // 6144 halves = 12 KB per e-chunk
#define CHUNK_UINT4  (CHUNK_HALVES * 2 / 16) // 768 uint4 per e-chunk

static __device__ __forceinline__ v16h mk16(v8h lo, v8h hi) {
  v16h r;
#pragma unroll
  for (int i = 0; i < 8; ++i) { r[i] = lo[i]; r[8 + i] = hi[i]; }
  return r;
}

static __device__ __forceinline__ v8f wmma16(v16h a, v16h b, v8f c) {
  // D = A(16x32 f16) * B(32x16 f16) + C(16x16 f32)
  return __builtin_amdgcn_wmma_f32_16x16x32_f16(false, a, false, b, (short)0, c,
                                                false, false);
}

// ---------------------------------------------------------------------------
// Kernel 0 (one-time, 192 KB output): convert Wq/Wk/Wv fp32 -> f16 and
// scatter into exact WMMA B-fragment order:
//   Wf[cc(16)][p(3)][t(4)][lane(32)][j(16)]
// so each block's per-chunk staging becomes a contiguous 12 KB vector copy.
// ---------------------------------------------------------------------------
__global__ __launch_bounds__(256) void wfrag_prep_kernel(
    const float* __restrict__ Wq, const float* __restrict__ Wk,
    const float* __restrict__ Wv, _Float16* __restrict__ Wf)
{
  int gid = blockIdx.x * 256 + threadIdx.x;       // 0 .. 3*512*64-1
  if (gid >= 3 * EMB * HD) return;
  int p   = gid / (EMB * HD);
  int rem = gid % (EMB * HD);
  const float* w = (p == 0) ? Wq : (p == 1) ? Wk : Wv;
  float val = w[rem];                              // coalesced read
  int e  = rem >> 6, d  = rem & 63;
  int cc = e >> 5,   er = e & 31;                  // e-chunk of 32
  int h2 = (er >> 3) & 1, j = (er & 7) | ((er >> 4) << 3); // table order: half h -> K 8h..8h+7, 16+8h..16+8h+7
  int t  = d >> 4,   l2 = d & 15;                  // n-tile, column in tile
  Wf[(size_t)(((cc * 12 + p * 4 + t) * 32) + (h2 * 16 + l2)) * 16 + j] =
      (_Float16)val;
}

// ---------------------------------------------------------------------------
// Kernel 1: fused QKV projection.  One pass over x: each A-fragment of x
// feeds 12 WMMAs (3 projections x 4 n-tiles).  The current e-chunk of the
// pre-swizzled weights is staged by a straight uint4 copy into a
// double-buffered 2 x 12 KB LDS region (copy of chunk cc+1 overlaps the
// WMMAs of chunk cc; one barrier per iteration).
// Q,K stored row-major f16 [nh][s][d]; V stored transposed f16 [nh][d][s].
// ---------------------------------------------------------------------------
__global__ __launch_bounds__(256) void qkv_fused_f16wmma_kernel(
    const float* __restrict__ x,
    const _Float16* __restrict__ Wf,
    const float* __restrict__ bq, const float* __restrict__ bk,
    const float* __restrict__ bv,
    _Float16* __restrict__ Qh, _Float16* __restrict__ Kh,
    _Float16* __restrict__ Vt)
{
  __shared__ __align__(16) _Float16 wfrag[2][CHUNK_HALVES];   // 24 KB

  const int tid  = threadIdx.x;
  const int wave = tid >> 5;
  const int lane = tid & 31;
  const int hi   = lane >> 4;
  const int ln   = lane & 15;

  const int nh    = blockIdx.y;           // n*8 + h
  const int n     = nh >> 3;
  const int h     = nh & 7;
  const int stile = blockIdx.x;           // 128-row S tile
  const int row0  = stile * 128 + wave * 16;

  // A-layout: lane holds row m = ln (both lane halves), K offset split by hi.
  const float* xrow = x + (((size_t)n * SEQ + row0 + ln) * NHEAD + h) * EMB;

  const uint4* wsrc = (const uint4*)Wf;

  // Preload chunk 0.
  {
    uint4* d = (uint4*)&wfrag[0][0];
#pragma unroll
    for (int i = 0; i < 3; ++i) d[tid + i * 256] = wsrc[tid + i * 256];
  }
  __syncthreads();

  v8f acc[3][4] = {};

  for (int cc = 0; cc < 16; ++cc) {
    const int cur = cc & 1;
    if (cc + 1 < 16) {                    // stage next chunk into other buffer
      uint4* d = (uint4*)&wfrag[cur ^ 1][0];
      const uint4* s = wsrc + (size_t)(cc + 1) * CHUNK_UINT4;
#pragma unroll
      for (int i = 0; i < 3; ++i) d[tid + i * 256] = s[tid + i * 256];
    }

    // A fragment: row ln, e = cc*32 + 8*hi + {0..7} and +16..+23
    const float* xa = xrow + cc * 32 + 8 * hi;
    if (cc + 1 < 16) __builtin_prefetch(xa + 32, 0, 3);
    v4f f0 = *(const v4f*)(xa);
    v4f f1 = *(const v4f*)(xa + 4);
    v4f f2 = *(const v4f*)(xa + 16);
    v4f f3 = *(const v4f*)(xa + 20);
    v16h a;
#pragma unroll
    for (int i = 0; i < 4; ++i) {
      a[i]      = (_Float16)f0[i];
      a[4 + i]  = (_Float16)f1[i];
      a[8 + i]  = (_Float16)f2[i];
      a[12 + i] = (_Float16)f3[i];
    }

#pragma unroll
    for (int p = 0; p < 3; ++p) {
#pragma unroll
      for (int t = 0; t < 4; ++t) {
        v16h b = *(const v16h*)&wfrag[cur][((p * 4 + t) * 32 + lane) * 16];
        acc[p][t] = wmma16(a, b, acc[p][t]);
      }
    }
    __syncthreads();   // next buffer fully written; current reads retired
  }

  // Bias + store.  C/D layout: VGPR r -> row r + 8*hi, lane%16 -> column.
  const float* Bs[3] = {bq, bk, bv};
#pragma unroll
  for (int p = 0; p < 3; ++p) {
    const float* bias = Bs[p];
    if (p < 2) {
      _Float16* outp = (p == 0 ? Qh : Kh) + (size_t)nh * SEQ * HD;
#pragma unroll
      for (int t = 0; t < 4; ++t) {
        float bb = bias[t * 16 + ln];
#pragma unroll
        for (int r = 0; r < 8; ++r) {
          int row = row0 + r + 8 * hi;
          outp[(size_t)row * HD + t * 16 + ln] = (_Float16)(acc[p][t][r] + bb);
        }
      }
    } else {
      // V transposed: Vt[nh][d][s]; per lane rows r are contiguous in s.
#pragma unroll
      for (int t = 0; t < 4; ++t) {
        float bb = bias[t * 16 + ln];
        v8h pk;
#pragma unroll
        for (int r = 0; r < 8; ++r) pk[r] = (_Float16)(acc[p][t][r] + bb);
        int d = t * 16 + ln;
        *(v8h*)&Vt[((size_t)nh * HD + d) * SEQ + row0 + 8 * hi] = pk;
      }
    }
  }
}

// ---------------------------------------------------------------------------
// Kernel 2: flash attention.  Each wave owns 16 query rows, streams key
// blocks of 32 with an online softmax.  All matmuls are f16 WMMA; the P
// tile round-trips through a per-wave padded LDS region to convert from
// C/D layout to A-fragment layout.  Q/K/Vt (24 MB f16) are L2-resident.
// ---------------------------------------------------------------------------
__global__ __launch_bounds__(256) void flash_attn_f16wmma_kernel(
    const _Float16* __restrict__ Qh, const _Float16* __restrict__ Kh,
    const _Float16* __restrict__ Vt, float* __restrict__ out)
{
  __shared__ _Float16 plds[8 * 16 * 40];   // 8 waves x (16 rows x 32 cols, +8 pad)

  const int tid  = threadIdx.x;
  const int wave = tid >> 5;
  const int lane = tid & 31;
  const int hi   = lane >> 4;
  const int ln   = lane & 15;

  const int nh    = blockIdx.y;
  const int n     = nh >> 3;
  const int h     = nh & 7;
  const int qtile = blockIdx.x;
  const int row0  = qtile * 128 + wave * 16;

  // Q fragments (A layout, row = ln), kept in registers for the whole loop.
  const _Float16* qrow = Qh + ((size_t)nh * SEQ + row0 + ln) * HD;
  v16h aq[2];
#pragma unroll
  for (int c = 0; c < 2; ++c) {
    v8h lo = *(const v8h*)(qrow + c * 32 + 16 * hi);
    v8h hh = *(const v8h*)(qrow + c * 32 + 16 * hi + 8);
    aq[c] = mk16(lo, hh);
  }

  const _Float16* kbase = Kh + (size_t)nh * SEQ * HD;
  const _Float16* vbase = Vt + (size_t)nh * HD * SEQ;
  _Float16* prow = &plds[(wave * 16) * 40];

  float mi[8], li[8];
#pragma unroll
  for (int r = 0; r < 8; ++r) { mi[r] = -__builtin_inff(); li[r] = 0.f; }
  v8f o[4] = {};
  const v8f zero = {};

  const float inv_scale = 0.044194173824159216f;   // 1/sqrt(EMB=512)

  for (int kb = 0; kb < SEQ / 32; ++kb) {
    if (kb + 1 < SEQ / 32) {   // prefetch next K/V tiles toward L2/WGP$
      __builtin_prefetch(kbase + (size_t)((kb + 1) * 32 + lane) * HD, 0, 3);
      __builtin_prefetch(vbase + (size_t)lane * SEQ + (kb + 1) * 32, 0, 3);
    }

    // Scores S = Q * K^T (key rows are B columns -> contiguous row loads).
    v8f s[2];
#pragma unroll
    for (int t = 0; t < 2; ++t) {
      const _Float16* krow =
          kbase + (size_t)(kb * 32 + t * 16 + ln) * HD + 16 * hi;
      v16h b0 = mk16(*(const v8h*)(krow),      *(const v8h*)(krow + 8));
      v16h b1 = mk16(*(const v8h*)(krow + 32), *(const v8h*)(krow + 40));
      v8f acc = wmma16(aq[0], b0, zero);
      s[t] = wmma16(aq[1], b1, acc);
    }
#pragma unroll
    for (int t = 0; t < 2; ++t) s[t] *= inv_scale;

    // Online softmax: one row lives in one 16-lane group per VGPR index r.
#pragma unroll
    for (int r = 0; r < 8; ++r) {
      float vmx = fmaxf(s[0][r], s[1][r]);
#pragma unroll
      for (int off = 1; off < 16; off <<= 1)
        vmx = fmaxf(vmx, __shfl_xor(vmx, off, 32));
      float mnew = fmaxf(mi[r], vmx);
      float sf   = __expf(mi[r] - mnew);
      float p0   = __expf(s[0][r] - mnew);
      float p1   = __expf(s[1][r] - mnew);
      float rs   = p0 + p1;
#pragma unroll
      for (int off = 1; off < 16; off <<= 1)
        rs += __shfl_xor(rs, off, 32);
      li[r] = li[r] * sf + rs;
      mi[r] = mnew;
#pragma unroll
      for (int t = 0; t < 4; ++t) o[t][r] *= sf;
      // Stage P (C-layout element (r+8*hi, col)) into LDS row-major f16.
      int prw = (r + 8 * hi) * 40;
      prow[prw + ln]      = (_Float16)p0;
      prow[prw + 16 + ln] = (_Float16)p1;
    }

    // Re-read P in A-fragment layout (same-wave LDS ops are in order).
    const _Float16* pr = &prow[ln * 40 + 16 * hi];
    v16h ap = mk16(*(const v8h*)(pr), *(const v8h*)(pr + 8));

    // O += P * V  (Vt rows give contiguous B-fragment loads).
#pragma unroll
    for (int t = 0; t < 4; ++t) {
      const _Float16* vr =
          vbase + (size_t)(t * 16 + ln) * SEQ + kb * 32 + 16 * hi;
      v16h bv = mk16(*(const v8h*)(vr), *(const v8h*)(vr + 8));
      o[t] = wmma16(ap, bv, o[t]);
    }
  }

  // Normalize and write out[n, q, h, d] fp32.
#pragma unroll
  for (int r = 0; r < 8; ++r) {
    float inv_l = 1.f / li[r];
    int row = row0 + r + 8 * hi;
    float* orow = out + (((size_t)n * SEQ + row) * NHEAD + h) * HD;
#pragma unroll
    for (int t = 0; t < 4; ++t)
      orow[t * 16 + ln] = o[t][r] * inv_l;
  }
}

// ---------------------------------------------------------------------------
extern "C" void kernel_launch(void* const* d_in, const int* in_sizes, int n_in,
                              void* d_out, int out_size, void* d_ws, size_t ws_size,
                              hipStream_t stream) {
  const float* x  = (const float*)d_in[0];
  const float* Wq = (const float*)d_in[1];
  const float* bq = (const float*)d_in[2];
  const float* Wk = (const float*)d_in[3];
  const float* bk = (const float*)d_in[4];
  const float* Wv = (const float*)d_in[5];
  const float* bv = (const float*)d_in[6];
  float* out = (float*)d_out;

  const size_t per = (size_t)NB * NHEAD * SEQ * HD;  // 4,194,304 halves (8 MB)
  _Float16* Qh = (_Float16*)d_ws;
  _Float16* Kh = Qh + per;
  _Float16* Vt = Kh + per;
  _Float16* Wf = Vt + per;                            // 98,304 halves (192 KB)

  wfrag_prep_kernel<<<(3 * EMB * HD + 255) / 256, 256, 0, stream>>>(Wq, Wk, Wv,
                                                                    Wf);
  dim3 grid(SEQ / 128, NB * NHEAD);
  qkv_fused_f16wmma_kernel<<<grid, 256, 0, stream>>>(x, Wf, bq, bk, bv,
                                                     Qh, Kh, Vt);
  flash_attn_f16wmma_kernel<<<grid, 256, 0, stream>>>(Qh, Kh, Vt, out);
}
